// GATv2Regressor_51067161150030
// MI455X (gfx1250) — matrix operand order repaired and checked by reading, not compiled, not run on hardware
//
#include <hip/hip_runtime.h>
#include <hip/hip_bf16.h>
#include <math.h>

// ---------------------------------------------------------------------------
// GATv2 regressor for MI455X (gfx1250, wave32).
// Dense node transforms: LDS-staged f16 tiles -> v_wmma_f32_16x16x32_f16.
// Edge softmax/aggregation: float4 gathers + f32 atomics (L2-resident).
// ---------------------------------------------------------------------------

typedef __attribute__((ext_vector_type(16))) _Float16 v16h;
typedef __attribute__((ext_vector_type(8)))  _Float16 v8h;
typedef __attribute__((ext_vector_type(8)))  float    v8f;

#define NEG_SLOPE 0.2f

// ------------------------------ helpers ------------------------------------

__device__ inline void atomicMaxF(float* addr, float val) {
    unsigned int* ia = (unsigned int*)addr;
    unsigned int old = *ia;
    while (__uint_as_float(old) < val) {
        unsigned int assumed = old;
        old = atomicCAS(ia, assumed, __float_as_uint(val));
        if (old == assumed) break;
    }
}

__global__ void fill_kernel(float* p, float v, int n) {
    int t = blockIdx.x * blockDim.x + threadIdx.x;
    if (t < n) p[t] = v;
}

__global__ void cast_f32_to_f16(const float* __restrict__ src,
                                _Float16* __restrict__ dst, int n) {
    int t = blockIdx.x * blockDim.x + threadIdx.x;
    if (t < n) dst[t] = (_Float16)src[t];
}

// Pack [Wl | Wr] (each K x halfN, row-major f32) into a COLUMN-MAJOR f16 panel
// Bt[col*K + k], and [bl ; br] into one Ntot f32 bias vector.
__global__ void pack_weights_colmajor(const float* __restrict__ Wl, const float* __restrict__ Wr,
                                      const float* __restrict__ bl, const float* __restrict__ br,
                                      _Float16* __restrict__ wt, float* __restrict__ biasc,
                                      int K, int Ntot) {
    int t = blockIdx.x * blockDim.x + threadIdx.x;
    int halfN = Ntot >> 1;
    if (t < K * Ntot) {
        int k = t / Ntot;
        int j = t - k * Ntot;
        float v = (j < halfN) ? Wl[k * halfN + j] : Wr[k * halfN + (j - halfN)];
        wt[(size_t)j * K + k] = (_Float16)v;       // column-major
    }
    if (t < Ntot) {
        biasc[t] = (t < halfN) ? bl[t] : br[t - halfN];
    }
}

// ------------------------------ WMMA GEMM ----------------------------------
// C[M x Ncols] = A[M x K](f16, row-major) * Bt[Ncols x K](f16, col-major) + bias
// Block = 4 waves. Block covers one 16-row M tile and a 64-column N panel.
// A tile (16 x K) and B panel (64 x K) are staged in LDS, then each wave
// runs a fully unrolled chain of 16x16x32 f16 WMMAs out of LDS.
// Requires M % 16 == 0, K % 32 == 0.

template <int K>
__global__ void wmma_gemm_f16(const _Float16* __restrict__ A,
                              const _Float16* __restrict__ Bt,
                              const float* __restrict__ bias,
                              float* __restrict__ C,
                              int M, int Ncols) {
    __shared__ _Float16 ash[16 * K];   // A tile, row-major (row*K + k)
    __shared__ _Float16 bsh[64 * K];   // B panel, col-major (localcol*K + k)

    const int tid   = threadIdx.x;          // 0..127
    const int wave  = tid >> 5;
    const int lane  = tid & 31;
    const int mtile = blockIdx.x * 16;
    const int nbase = blockIdx.y * 64;      // first column of this block's panel

    // --- cooperative stage: A tile (16*K halves, contiguous rows) ---
    {
        const _Float16* asrc = A + (size_t)mtile * K;
        for (int idx = tid; idx < (16 * K) / 8; idx += 128) {
            *(v8h*)(ash + idx * 8) = *(const v8h*)(asrc + idx * 8);
        }
        // --- B panel: 64 contiguous columns of col-major Bt ---
        const _Float16* bsrc = Bt + (size_t)nbase * K;
        for (int idx = tid; idx < (64 * K) / 8; idx += 128) {
            *(v8h*)(bsh + idx * 8) = *(const v8h*)(bsrc + idx * 8);
        }
    }
    __syncthreads();

    const int arow  = lane & 15;             // A row held by this lane
    const int kgrpA = (lane >> 4) * 8;       // lanes 0-15 -> K+0, 16-31 -> K+8
    const int bcolL = wave * 16 + (lane & 15);  // local B column
    const int kgrpB = (lane >> 4) * 8;       // lanes 0-15 -> K+0, 16-31 -> K+8

    v8f acc = {};
    #pragma unroll
    for (int k0 = 0; k0 < K; k0 += 32) {
        // A frag: elems 0..7 -> k0+kgrpA+e ; elems 8..15 -> k0+kgrpA+16+e
        union { v16h v; v8h h[2]; } af;
        const _Float16* ap = ash + arow * K + k0 + kgrpA;
        af.h[0] = *(const v8h*)(ap);
        af.h[1] = *(const v8h*)(ap + 16);

        // B frag: elems 0..7 -> k0+kgrpB+e ; elems 8..15 -> k0+kgrpB+16+e
        union { v16h v; v8h h[2]; } bf;
        const _Float16* bp = bsh + bcolL * K + k0 + kgrpB;
        bf.h[0] = *(const v8h*)(bp);
        bf.h[1] = *(const v8h*)(bp + 16);

        acc = __builtin_amdgcn_wmma_f32_16x16x32_f16(
            false, af.v, false, bf.v, (short)0, acc, false, false);
    }

    // C/D layout: lanes 0-15: VGPR r -> M=r ; lanes 16-31: M=8+r ; N = lane&15
    const int ncol    = nbase + bcolL;
    const int rowbase = mtile + ((lane >> 4) << 3);
    const float bcol  = bias[ncol];
    #pragma unroll
    for (int r = 0; r < 8; ++r) {
        C[(size_t)(rowbase + r) * Ncols + ncol] = acc[r] + bcol;
    }
}

// ------------------------------ edge phase ---------------------------------
// xlr: N x ld panel, columns [0, ld/2) = xl, [ld/2, ld) = xr. C=32 per head.

__global__ void edge_score_max(const int* __restrict__ srcA, const int* __restrict__ dstA,
                               int Eorig, int Etot,
                               const float* __restrict__ xlr, int ld,
                               const float* __restrict__ att, int H,
                               float* __restrict__ escore, float* __restrict__ nodemax) {
    int t = blockIdx.x * blockDim.x + threadIdx.x;
    if (t >= Etot * H) return;
    int e = t / H, h = t - e * H;
    int s, d;
    if (e < Eorig) { s = srcA[e]; d = dstA[e]; } else { s = d = e - Eorig; }
    int half = ld >> 1;
    const float4* xl4 = (const float4*)(xlr + (size_t)s * ld + h * 32);
    const float4* xr4 = (const float4*)(xlr + (size_t)d * ld + half + h * 32);
    const float4* at4 = (const float4*)(att + h * 32);
    float sc = 0.f;
    #pragma unroll
    for (int i = 0; i < 8; ++i) {
        float4 a = xl4[i], b = xr4[i], w = at4[i];
        float v0 = a.x + b.x, v1 = a.y + b.y, v2 = a.z + b.z, v3 = a.w + b.w;
        v0 = (v0 > 0.f) ? v0 : NEG_SLOPE * v0;
        v1 = (v1 > 0.f) ? v1 : NEG_SLOPE * v1;
        v2 = (v2 > 0.f) ? v2 : NEG_SLOPE * v2;
        v3 = (v3 > 0.f) ? v3 : NEG_SLOPE * v3;
        sc += v0 * w.x + v1 * w.y + v2 * w.z + v3 * w.w;
    }
    escore[t] = sc;
    atomicMaxF(&nodemax[(size_t)d * H + h], sc);
}

__global__ void edge_exp_sum(const int* __restrict__ dstA,
                             int Eorig, int Etot, int H,
                             float* __restrict__ escore,
                             const float* __restrict__ nodemax,
                             float* __restrict__ nodesum) {
    int t = blockIdx.x * blockDim.x + threadIdx.x;
    if (t >= Etot * H) return;
    int e = t / H, h = t - e * H;
    int d = (e < Eorig) ? dstA[e] : (e - Eorig);
    float w = __expf(escore[t] - nodemax[(size_t)d * H + h]);
    escore[t] = w;
    atomicAdd(&nodesum[(size_t)d * H + h], w);
}

__global__ void edge_aggregate(const int* __restrict__ srcA, const int* __restrict__ dstA,
                               int Eorig, int Etot,
                               const float* __restrict__ xlr, int ld,
                               const float* __restrict__ escore,
                               const float* __restrict__ nodesum, int H,
                               float* __restrict__ out) {
    int t = blockIdx.x * blockDim.x + threadIdx.x;
    if (t >= Etot * H) return;
    int e = t / H, h = t - e * H;
    int s, d;
    if (e < Eorig) { s = srcA[e]; d = dstA[e]; } else { s = d = e - Eorig; }
    float w = escore[t] / (nodesum[(size_t)d * H + h] + 1e-16f);
    const float4* xl4 = (const float4*)(xlr + (size_t)s * ld + h * 32);
    float* o = out + (size_t)d * (H * 32) + h * 32;
    #pragma unroll
    for (int i = 0; i < 8; ++i) {
        float4 a = xl4[i];
        atomicAdd(&o[4 * i + 0], w * a.x);
        atomicAdd(&o[4 * i + 1], w * a.y);
        atomicAdd(&o[4 * i + 2], w * a.z);
        atomicAdd(&o[4 * i + 3], w * a.w);
    }
}

// bias + relu; optionally also emit an f16 copy for the next WMMA layer.
__global__ void bias_relu_cast(const float* __restrict__ in, const float* __restrict__ bias,
                               float* __restrict__ outf, _Float16* __restrict__ outh,
                               int n, int cols) {
    int t = blockIdx.x * blockDim.x + threadIdx.x;
    if (t >= n) return;
    int j = t % cols;
    float v = in[t] + bias[j];
    v = (v > 0.f) ? v : 0.f;
    if (outf) outf[t] = v;
    if (outh) outh[t] = (_Float16)v;
}

// ------------------------------ pooling + head -----------------------------

__global__ void gate_compute(const float* __restrict__ h2,
                             const float* __restrict__ Wg1, const float* __restrict__ bg1,
                             const float* __restrict__ Wg2, const float* __restrict__ bg2,
                             const int* __restrict__ batch,
                             float* __restrict__ gate, float* __restrict__ gmax, int N) {
    int n = blockIdx.x * blockDim.x + threadIdx.x;
    if (n >= N) return;
    float hrow[32];
    const float4* h4 = (const float4*)(h2 + (size_t)n * 32);
    #pragma unroll
    for (int i = 0; i < 8; ++i) {
        float4 v = h4[i];
        hrow[4 * i + 0] = v.x; hrow[4 * i + 1] = v.y;
        hrow[4 * i + 2] = v.z; hrow[4 * i + 3] = v.w;
    }
    float acc = bg2[0];
    for (int c = 0; c < 32; ++c) {
        float tv = bg1[c];
        #pragma unroll
        for (int k = 0; k < 32; ++k) tv += hrow[k] * Wg1[k * 32 + c];
        tv = (tv > 0.f) ? tv : 0.f;
        acc += tv * Wg2[c];
    }
    gate[n] = acc;
    atomicMaxF(&gmax[batch[n]], acc);
}

__global__ void gate_exp_sum(float* __restrict__ gate, const float* __restrict__ gmax,
                             float* __restrict__ gsum, const int* __restrict__ batch, int N) {
    int n = blockIdx.x * blockDim.x + threadIdx.x;
    if (n >= N) return;
    int b = batch[n];
    float w = __expf(gate[n] - gmax[b]);
    gate[n] = w;
    atomicAdd(&gsum[b], w);
}

__global__ void pool_scatter(const float* __restrict__ gate, const float* __restrict__ gsum,
                             const float* __restrict__ h2, const int* __restrict__ batch,
                             float* __restrict__ p, int N) {
    int n = blockIdx.x * blockDim.x + threadIdx.x;
    if (n >= N) return;
    int b = batch[n];
    float a = gate[n] / (gsum[b] + 1e-16f);
    const float4* h4 = (const float4*)(h2 + (size_t)n * 32);
    #pragma unroll
    for (int i = 0; i < 8; ++i) {
        float4 v = h4[i];
        atomicAdd(&p[(size_t)b * 32 + 4 * i + 0], a * v.x);
        atomicAdd(&p[(size_t)b * 32 + 4 * i + 1], a * v.y);
        atomicAdd(&p[(size_t)b * 32 + 4 * i + 2], a * v.z);
        atomicAdd(&p[(size_t)b * 32 + 4 * i + 3], a * v.w);
    }
}

__global__ void head_mlp(const float* __restrict__ p,
                         const float* __restrict__ W1, const float* __restrict__ b1,
                         const float* __restrict__ W2, const float* __restrict__ b2,
                         float* __restrict__ out, int G) {
    int g = blockIdx.x * blockDim.x + threadIdx.x;
    if (g >= G) return;
    float prow[32];
    #pragma unroll
    for (int k = 0; k < 32; ++k) prow[k] = p[(size_t)g * 32 + k];
    float acc = b2[0];
    for (int c = 0; c < 32; ++c) {
        float tv = b1[c];
        #pragma unroll
        for (int k = 0; k < 32; ++k) tv += prow[k] * W1[k * 32 + c];
        tv = (tv > 0.f) ? tv : 0.f;
        acc += tv * W2[c];
    }
    out[g] = acc;
}

// ------------------------------ launch -------------------------------------

static inline size_t align256(size_t x) { return (x + 255) & ~(size_t)255; }
static inline int ceil_div(int a, int b) { return (a + b - 1) / b; }

extern "C" void kernel_launch(void* const* d_in, const int* in_sizes, int n_in,
                              void* d_out, int out_size, void* d_ws, size_t ws_size,
                              hipStream_t stream) {
    const float* x     = (const float*)d_in[0];
    const int*   ei    = (const int*)d_in[1];
    const int*   batch = (const int*)d_in[2];
    const float* Wl1  = (const float*)d_in[3];
    const float* bl1  = (const float*)d_in[4];
    const float* Wr1  = (const float*)d_in[5];
    const float* br1  = (const float*)d_in[6];
    const float* att1 = (const float*)d_in[7];
    const float* bias1= (const float*)d_in[8];
    const float* Wl2  = (const float*)d_in[9];
    const float* bl2  = (const float*)d_in[10];
    const float* Wr2  = (const float*)d_in[11];
    const float* br2  = (const float*)d_in[12];
    const float* att2 = (const float*)d_in[13];
    const float* bias2= (const float*)d_in[14];
    const float* Wg1  = (const float*)d_in[15];
    const float* bg1  = (const float*)d_in[16];
    const float* Wg2  = (const float*)d_in[17];
    const float* bg2  = (const float*)d_in[18];
    const float* W1   = (const float*)d_in[19];
    const float* b1   = (const float*)d_in[20];
    const float* W2   = (const float*)d_in[21];
    const float* b2   = (const float*)d_in[22];

    const int F = 128, H1 = 2;
    const int N = in_sizes[0] / F;
    const int E = in_sizes[1] / 2;
    const int Etot = E + N;           // with self-loops
    const int G = out_size;
    const int* srcA = ei;
    const int* dstA = ei + E;

    // workspace carve
    char* wp = (char*)d_ws;
    auto take = [&](size_t bytes) -> void* { void* p = (void*)wp; wp += align256(bytes); return p; };
    _Float16* xh     = (_Float16*)take((size_t)N * F * 2);
    _Float16* wt1    = (_Float16*)take((size_t)F * 128 * 2);   // col-major 128x128
    float*    biasc1 = (float*)   take(128 * 4);
    float*    xlr1   = (float*)   take((size_t)N * 128 * 4);
    float*    esc1   = (float*)   take((size_t)Etot * H1 * 4);
    float*    nmax1  = (float*)   take((size_t)N * H1 * 4);
    float*    nsum1  = (float*)   take((size_t)N * H1 * 4);
    float*    out1   = (float*)   take((size_t)N * 64 * 4);
    _Float16* h1h    = (_Float16*)take((size_t)N * 64 * 2);
    _Float16* wt2    = (_Float16*)take((size_t)64 * 64 * 2);   // col-major 64x64
    float*    biasc2 = (float*)   take(64 * 4);
    float*    xlr2   = (float*)   take((size_t)N * 64 * 4);
    float*    esc2   = (float*)   take((size_t)Etot * 4);
    float*    nmax2  = (float*)   take((size_t)N * 4);
    float*    nsum2  = (float*)   take((size_t)N * 4);
    float*    out2   = (float*)   take((size_t)N * 32 * 4);
    float*    h2     = (float*)   take((size_t)N * 32 * 4);
    float*    gate   = (float*)   take((size_t)N * 4);
    float*    gmax   = (float*)   take((size_t)G * 4);
    float*    gsum   = (float*)   take((size_t)G * 4);
    float*    pbuf   = (float*)   take((size_t)G * 32 * 4);

    const int T = 256;

    // prep: f16 activations + packed (transposed) weight panels
    cast_f32_to_f16<<<ceil_div(N * F, T), T, 0, stream>>>(x, xh, N * F);
    pack_weights_colmajor<<<ceil_div(F * 128, T), T, 0, stream>>>(Wl1, Wr1, bl1, br1, wt1, biasc1, F, 128);
    pack_weights_colmajor<<<ceil_div(64 * 64, T), T, 0, stream>>>(Wl2, Wr2, bl2, br2, wt2, biasc2, 64, 64);

    // init accumulators
    fill_kernel<<<ceil_div(N * H1, T), T, 0, stream>>>(nmax1, -1e30f, N * H1);
    fill_kernel<<<ceil_div(N * H1, T), T, 0, stream>>>(nsum1, 0.f, N * H1);
    fill_kernel<<<ceil_div(N * 64, T), T, 0, stream>>>(out1, 0.f, N * 64);
    fill_kernel<<<ceil_div(N, T), T, 0, stream>>>(nmax2, -1e30f, N);
    fill_kernel<<<ceil_div(N, T), T, 0, stream>>>(nsum2, 0.f, N);
    fill_kernel<<<ceil_div(N * 32, T), T, 0, stream>>>(out2, 0.f, N * 32);
    fill_kernel<<<ceil_div(G, T), T, 0, stream>>>(gmax, -1e30f, G);
    fill_kernel<<<ceil_div(G, T), T, 0, stream>>>(gsum, 0.f, G);
    fill_kernel<<<ceil_div(G * 32, T), T, 0, stream>>>(pbuf, 0.f, G * 32);

    // ---- layer 1: xlr1 = xh @ [Wl1|Wr1] + [bl1;br1]  (N x 128) ----
    wmma_gemm_f16<128><<<dim3(N / 16, 2), 128, 0, stream>>>(xh, wt1, biasc1, xlr1, N, 128);
    edge_score_max<<<ceil_div(Etot * H1, T), T, 0, stream>>>(srcA, dstA, E, Etot, xlr1, 128, att1, H1, esc1, nmax1);
    edge_exp_sum<<<ceil_div(Etot * H1, T), T, 0, stream>>>(dstA, E, Etot, H1, esc1, nmax1, nsum1);
    edge_aggregate<<<ceil_div(Etot * H1, T), T, 0, stream>>>(srcA, dstA, E, Etot, xlr1, 128, esc1, nsum1, H1, out1);
    bias_relu_cast<<<ceil_div(N * 64, T), T, 0, stream>>>(out1, bias1, nullptr, h1h, N * 64, 64);

    // ---- layer 2: xlr2 = h1 @ [Wl2|Wr2] + [bl2;br2]  (N x 64) ----
    wmma_gemm_f16<64><<<dim3(N / 16, 1), 128, 0, stream>>>(h1h, wt2, biasc2, xlr2, N, 64);
    edge_score_max<<<ceil_div(Etot, T), T, 0, stream>>>(srcA, dstA, E, Etot, xlr2, 64, att2, 1, esc2, nmax2);
    edge_exp_sum<<<ceil_div(Etot, T), T, 0, stream>>>(dstA, E, Etot, 1, esc2, nmax2, nsum2);
    edge_aggregate<<<ceil_div(Etot, T), T, 0, stream>>>(srcA, dstA, E, Etot, xlr2, 64, esc2, nsum2, 1, out2);
    bias_relu_cast<<<ceil_div(N * 32, T), T, 0, stream>>>(out2, bias2, h2, nullptr, N * 32, 32);

    // ---- global attention pooling + head ----
    gate_compute<<<ceil_div(N, T), T, 0, stream>>>(h2, Wg1, bg1, Wg2, bg2, batch, gate, gmax, N);
    gate_exp_sum<<<ceil_div(N, T), T, 0, stream>>>(gate, gmax, gsum, batch, N);
    pool_scatter<<<ceil_div(N, T), T, 0, stream>>>(gate, gsum, h2, batch, pbuf, N);
    head_mlp<<<ceil_div(G, 64), 64, 0, stream>>>(pbuf, W1, b1, W2, b2, (float*)d_out, G);
}
